// ConvLSTM2d_model_59098749993406
// MI455X (gfx1250) — matrix operand order repaired and checked by reading, not compiled, not run on hardware
//
#include <hip/hip_runtime.h>
#include <hip/hip_bf16.h>
#include <math.h>

#define BATCH 16
#define TT    10
#define HH    64
#define WW    64
#define FF    64
#define CG    256   // 4*F gate channels

typedef _Float16 v16h __attribute__((ext_vector_type(16)));
typedef _Float16 v8h  __attribute__((ext_vector_type(8)));
typedef float    v8f  __attribute__((ext_vector_type(8)));
typedef int      v4i  __attribute__((vector_size(16)));

#define AS1 __attribute__((address_space(1)))
#define AS3 __attribute__((address_space(3)))

#if defined(__has_builtin)
#if __has_builtin(__builtin_amdgcn_global_load_async_to_lds_b128)
#define HAVE_ASYNC_LDS 1
#endif
#if __has_builtin(__builtin_amdgcn_s_wait_asynccnt)
#define HAVE_WAIT_ASYNC 1
#endif
#endif

static __device__ __forceinline__ void async_copy16(const _Float16* gp,
                                                    _Float16* lp) {
#if defined(HAVE_ASYNC_LDS)
  // GLOBAL_LOAD_ASYNC_TO_LDS_B128: per-lane 16B global -> LDS, ASYNCcnt
  __builtin_amdgcn_global_load_async_to_lds_b128((AS1 v4i*)gp, (AS3 v4i*)lp, 0,
                                                 0);
#else
  *(v8h*)lp = *(const v8h*)gp;
#endif
}

static __device__ __forceinline__ void wait_async_lds() {
#if defined(HAVE_ASYNC_LDS)
#if defined(HAVE_WAIT_ASYNC)
  __builtin_amdgcn_s_wait_asynccnt(0);
#else
  asm volatile("s_wait_asynccnt 0" ::: "memory");
#endif
#endif
}

static __device__ __forceinline__ v8f wmma16(v16h a, v16h b, v8f c) {
  // D = A(16x32 f16) * B(32x16 f16) + C(16x16 f32)
  return __builtin_amdgcn_wmma_f32_16x16x32_f16(false, a, false, b, (short)0, c,
                                                false, false);
}

static __device__ __forceinline__ float hsig(float x) {
  return fminf(fmaxf(0.2f * x + 0.5f, 0.f), 1.f);
}

// A-fragment (16x32 f16): lane r holds row M=r; K layout per doc:
// elem j<8  -> K = 8*hf + j         (load 8 f16 at p + 8*hf)
// elem j>=8 -> K = 16 + 8*hf + j-8  (load 8 f16 at p + 16 + 8*hf)
static __device__ __forceinline__ v16h ld_afrag(const _Float16* p, int hf) {
  union { v16h v; v8h h[2]; } u;
  u.h[0] = *(const v8h*)(p + 8 * hf);
  u.h[1] = *(const v8h*)(p + 16 + 8 * hf);
  return u.v;
}

// B-fragment (32x16 f16): lane (r,hf) holds col N=r, K = 8*hf + j for j<8,
// K = 16 + 8*hf + j-8 for j>=8 (weights stored transposed [cout][cin]).
static __device__ __forceinline__ v16h ld_bfrag(const _Float16* p, int hf) {
  union { v16h v; v8h h[2]; } u;
  u.h[0] = *(const v8h*)(p + 8 * hf);
  u.h[1] = *(const v8h*)(p + 16 + 8 * hf);
  return u.v;
}

// ---------------- utility kernels ----------------

__global__ void zero_f32_kernel(float* p, int n) {
  int i = blockIdx.x * 256 + threadIdx.x;
  if (i < n) p[i] = 0.f;
}
__global__ void zero_f16_kernel(_Float16* p, int n) {
  int i = blockIdx.x * 256 + threadIdx.x;
  if (i < n) p[i] = (_Float16)0.f;
}

// HWIO (taps,cin,cout) f32  ->  [tap][cout][cin] f16 (B-fragment friendly)
__global__ void repack_oi_kernel(const float* __restrict__ src,
                                 _Float16* __restrict__ dst,
                                 int taps, int cin, int cout) {
  int i = blockIdx.x * 256 + threadIdx.x;
  int n = taps * cin * cout;
  if (i >= n) return;
  int ci = i % cin;
  int rem = i / cin;
  int co = rem % cout;
  int tap = rem / cout;
  dst[i] = (_Float16)src[(tap * cin + ci) * cout + co];
}

// wx1 (25,1,256) f32 -> [cout][32] f16, K = tap index padded 25->32
__global__ void repack_x1_kernel(const float* __restrict__ wx1,
                                 _Float16* __restrict__ dst) {
  int i = blockIdx.x * 256 + threadIdx.x;
  if (i >= 256 * 32) return;
  int k = i & 31;
  int co = i >> 5;
  dst[i] = (k < 25) ? (_Float16)wx1[k * 256 + co] : (_Float16)0.f;
}

// ---------------- ConvLSTM layer 1 step (5x5, Cin_x=1) ----------------
// grid: 2048 blocks = B(16) * y(64) * xblock(2); block 256 = 8 wave32.
// Workgroup tile: 32 pixels (one row segment) x 256 gate channels.
__global__ __launch_bounds__(256) void lstm1_step_kernel(
    const float* __restrict__ img, int t,
    const _Float16* __restrict__ wtx,   // [256][32] f16
    const _Float16* __restrict__ wth,   // [25][256][64] f16
    const float* __restrict__ bias,     // [256]
    const float* __restrict__ gmm, const float* __restrict__ bta,
    const float* __restrict__ mu, const float* __restrict__ vr,
    const _Float16* __restrict__ h_in,  // [B][H][W][F] f16
    _Float16* __restrict__ h_out,
    float* __restrict__ c_state,        // d_out c0 region, f32
    float* __restrict__ h_final,        // d_out h0 region, f32
    _Float16* __restrict__ x1seq,       // [B][T][H][W][F] f16 (post-BN)
    int last) {
  __shared__ __align__(16) _Float16 sh[5][36][72];  // h tile, 72 = 64 + 8 pad
  __shared__ __align__(16) _Float16 sx[5][40];      // x tile
  __shared__ __align__(16) float sz[32][CG];        // gate pre-activations

  const int tid = threadIdx.x;
  const int blk = blockIdx.x;
  const int xb = blk & 1;
  const int y = (blk >> 1) & 63;
  const int b = blk >> 7;
  const int x0 = xb * 32;

  __builtin_prefetch(wth, 0, 1);

  // stage h tile (rows y-2..y+2, cols x0-2..x0+33, 64 ch) channel-contiguous.
  // 16B chunk per lane via async global->LDS DMA; halo chunks get DS zeros.
  {
    v8h z;
#pragma unroll
    for (int j = 0; j < 8; ++j) z[j] = (_Float16)0.f;
    for (int q = tid; q < 5 * 36 * 8; q += 256) {
      int site = q >> 3, j = q & 7;
      int row = site / 36, col = site % 36;
      int yy = y + row - 2, xx = x0 + col - 2;
      _Float16* lp = &sh[row][col][j * 8];
      if (yy >= 0 && yy < HH && xx >= 0 && xx < WW) {
        const _Float16* gp = &h_in[((b * HH + yy) * WW + xx) * FF + j * 8];
        async_copy16(gp, lp);
      } else {
        *(v8h*)lp = z;
      }
    }
  }
  for (int s = tid; s < 5 * 36; s += 256) {
    int row = s / 36, col = s % 36;
    int yy = y + row - 2, xx = x0 + col - 2;
    float v = 0.f;
    if (yy >= 0 && yy < HH && xx >= 0 && xx < WW)
      v = img[((b * TT + t) * HH + yy) * WW + xx];
    sx[row][col] = (_Float16)v;
  }
  wait_async_lds();
  __syncthreads();

  const int lane = tid & 31;
  const int hf = lane >> 4;
  const int r = lane & 15;
  const int n0 = (tid >> 5) * 32;  // this wave's gate-channel base

  v8f vzero;
#pragma unroll
  for (int j = 0; j < 8; ++j) vzero[j] = 0.f;
  v8f acc[2][2];
  acc[0][0] = vzero; acc[0][1] = vzero; acc[1][0] = vzero; acc[1][1] = vzero;

  // ---- input (x) contribution: one K=32 chunk, K == tap index ----
  {
    v16h af[2];
#pragma unroll
    for (int mf = 0; mf < 2; ++mf) {
      int m = mf * 16 + r;
      v16h a;
#pragma unroll
      for (int j = 0; j < 16; ++j) {
        int k = (j < 8) ? (8 * hf + j) : (8 + 8 * hf + j);
        _Float16 v = (_Float16)0.f;
        if (k < 25) v = sx[k / 5][k % 5 + m];
        a[j] = v;
      }
      af[mf] = a;
    }
    v16h bf[2];
#pragma unroll
    for (int nf = 0; nf < 2; ++nf)
      bf[nf] = ld_bfrag(&wtx[(n0 + nf * 16 + r) * 32], hf);
#pragma unroll
    for (int mf = 0; mf < 2; ++mf)
#pragma unroll
      for (int nf = 0; nf < 2; ++nf)
        acc[mf][nf] = wmma16(af[mf], bf[nf], acc[mf][nf]);
  }

  // ---- recurrent (h) contribution: 25 taps x 2 K-chunks of 32 ch ----
  for (int tap = 0; tap < 25; ++tap) {
    int dy = tap / 5, dx = tap % 5;
    const _Float16* wb = wth + tap * CG * FF;
#pragma unroll
    for (int kc = 0; kc < 2; ++kc) {
      int c0 = kc * 32;
      v16h af[2];
#pragma unroll
      for (int mf = 0; mf < 2; ++mf)
        af[mf] = ld_afrag(&sh[dy][dx + mf * 16 + r][c0], hf);
      v16h bf[2];
#pragma unroll
      for (int nf = 0; nf < 2; ++nf)
        bf[nf] = ld_bfrag(wb + (n0 + nf * 16 + r) * FF + c0, hf);
#pragma unroll
      for (int mf = 0; mf < 2; ++mf)
#pragma unroll
        for (int nf = 0; nf < 2; ++nf)
          acc[mf][nf] = wmma16(af[mf], bf[nf], acc[mf][nf]);
    }
  }

  // ---- spill gates to LDS (D layout: elem i -> M = 8*hf + i, N = r) ----
#pragma unroll
  for (int mf = 0; mf < 2; ++mf)
#pragma unroll
    for (int nf = 0; nf < 2; ++nf)
#pragma unroll
      for (int i = 0; i < 8; ++i)
        sz[mf * 16 + 8 * hf + i][n0 + nf * 16 + r] = acc[mf][nf][i];
  __syncthreads();

  // ---- LSTM update + BN epilogue ----
  for (int e = tid; e < 32 * FF; e += 256) {
    int pix = e >> 6, ch = e & 63;
    float iv = hsig(sz[pix][ch] + bias[ch]);
    float fv = hsig(sz[pix][64 + ch] + bias[64 + ch]);
    float gv = tanhf(sz[pix][128 + ch] + bias[128 + ch]);
    float ov = hsig(sz[pix][192 + ch] + bias[192 + ch]);
    int gi = ((b * HH + y) * WW + (x0 + pix)) * FF + ch;
    float c = fv * c_state[gi] + iv * gv;
    c_state[gi] = c;
    float h = ov * tanhf(c);
    h_out[gi] = (_Float16)h;
    float bn = gmm[ch] * (h - mu[ch]) * rsqrtf(vr[ch] + 1e-3f) + bta[ch];
    x1seq[(((b * TT + t) * HH + y) * WW + (x0 + pix)) * FF + ch] = (_Float16)bn;
    if (last) h_final[gi] = h;
  }
}

// ---------------- ConvLSTM layer 2 step (3x3, Cin_x=64) ----------------
__global__ __launch_bounds__(256) void lstm2_step_kernel(
    const _Float16* __restrict__ x1seq, int t,
    const _Float16* __restrict__ wtx,   // [9][256][64] f16
    const _Float16* __restrict__ wth,   // [9][256][64] f16
    const float* __restrict__ bias,
    const float* __restrict__ gmm, const float* __restrict__ bta,
    const float* __restrict__ mu, const float* __restrict__ vr,
    const _Float16* __restrict__ h_in, _Float16* __restrict__ h_out,
    float* __restrict__ c_state, float* __restrict__ h_final,
    _Float16* __restrict__ x2seq, int last) {
  __shared__ __align__(16) _Float16 sa[2][3][34][72];  // [x1, h] tiles
  __shared__ __align__(16) float sz[32][CG];

  const int tid = threadIdx.x;
  const int blk = blockIdx.x;
  const int xb = blk & 1;
  const int y = (blk >> 1) & 63;
  const int b = blk >> 7;
  const int x0 = xb * 32;

  __builtin_prefetch(wtx, 0, 1);
  __builtin_prefetch(wth, 0, 1);

  {
    v8h z;
#pragma unroll
    for (int j = 0; j < 8; ++j) z[j] = (_Float16)0.f;
    for (int q = tid; q < 2 * 3 * 34 * 8; q += 256) {
      int site = q >> 3, j = q & 7;
      int src = site / 102;
      int rem = site % 102;
      int row = rem / 34, col = rem % 34;
      int yy = y + row - 1, xx = x0 + col - 1;
      _Float16* lp = &sa[src][row][col][j * 8];
      if (yy >= 0 && yy < HH && xx >= 0 && xx < WW) {
        const _Float16* gp =
            (src == 0)
                ? &x1seq[((((size_t)(b * TT + t)) * HH + yy) * WW + xx) * FF +
                         j * 8]
                : &h_in[((b * HH + yy) * WW + xx) * FF + j * 8];
        async_copy16(gp, lp);
      } else {
        *(v8h*)lp = z;
      }
    }
  }
  wait_async_lds();
  __syncthreads();

  const int lane = tid & 31;
  const int hf = lane >> 4;
  const int r = lane & 15;
  const int n0 = (tid >> 5) * 32;

  v8f vzero;
#pragma unroll
  for (int j = 0; j < 8; ++j) vzero[j] = 0.f;
  v8f acc[2][2];
  acc[0][0] = vzero; acc[0][1] = vzero; acc[1][0] = vzero; acc[1][1] = vzero;

  for (int src = 0; src < 2; ++src) {
    const _Float16* wsel = (src == 0) ? wtx : wth;
    for (int tap = 0; tap < 9; ++tap) {
      int dy = tap / 3, dx = tap % 3;
      const _Float16* wb = wsel + tap * CG * FF;
#pragma unroll
      for (int kc = 0; kc < 2; ++kc) {
        int c0 = kc * 32;
        v16h af[2];
#pragma unroll
        for (int mf = 0; mf < 2; ++mf)
          af[mf] = ld_afrag(&sa[src][dy][dx + mf * 16 + r][c0], hf);
        v16h bf[2];
#pragma unroll
        for (int nf = 0; nf < 2; ++nf)
          bf[nf] = ld_bfrag(wb + (n0 + nf * 16 + r) * FF + c0, hf);
#pragma unroll
        for (int mf = 0; mf < 2; ++mf)
#pragma unroll
          for (int nf = 0; nf < 2; ++nf)
            acc[mf][nf] = wmma16(af[mf], bf[nf], acc[mf][nf]);
      }
    }
  }

#pragma unroll
  for (int mf = 0; mf < 2; ++mf)
#pragma unroll
    for (int nf = 0; nf < 2; ++nf)
#pragma unroll
      for (int i = 0; i < 8; ++i)
        sz[mf * 16 + 8 * hf + i][n0 + nf * 16 + r] = acc[mf][nf][i];
  __syncthreads();

  for (int e = tid; e < 32 * FF; e += 256) {
    int pix = e >> 6, ch = e & 63;
    float iv = hsig(sz[pix][ch] + bias[ch]);
    float fv = hsig(sz[pix][64 + ch] + bias[64 + ch]);
    float gv = tanhf(sz[pix][128 + ch] + bias[128 + ch]);
    float ov = hsig(sz[pix][192 + ch] + bias[192 + ch]);
    int gi = ((b * HH + y) * WW + (x0 + pix)) * FF + ch;
    float c = fv * c_state[gi] + iv * gv;
    c_state[gi] = c;
    float h = ov * tanhf(c);
    h_out[gi] = (_Float16)h;
    float bn = gmm[ch] * (h - mu[ch]) * rsqrtf(vr[ch] + 1e-3f) + bta[ch];
    x2seq[(((b * TT + t) * HH + y) * WW + (x0 + pix)) * FF + ch] = (_Float16)bn;
    if (last) h_final[gi] = h;
  }
}

// ---------------- Conv3D (1,3,3) head, Cin=129, Cout=1, sigmoid ----------------
__global__ __launch_bounds__(256) void head_kernel(
    const float* __restrict__ img, const _Float16* __restrict__ x1s,
    const _Float16* __restrict__ x2s, const float* __restrict__ w3,
    const float* __restrict__ b3, float* __restrict__ out) {
  int i = blockIdx.x * 256 + threadIdx.x;
  if (i >= BATCH * TT * HH * WW) return;
  int x = i & 63;
  int y = (i >> 6) & 63;
  int bt = i >> 12;  // b*T + t
  float acc = b3[0];
  for (int dy = -1; dy <= 1; ++dy) {
    int yy = y + dy;
    if (yy < 0 || yy >= HH) continue;
    for (int dx = -1; dx <= 1; ++dx) {
      int xx = x + dx;
      if (xx < 0 || xx >= WW) continue;
      const float* w = w3 + ((dy + 1) * 3 + (dx + 1)) * 129;
      int pix = (bt * HH + yy) * WW + xx;
      acc += w[0] * img[pix];
      const _Float16* p1 = x1s + (size_t)pix * FF;
      const _Float16* p2 = x2s + (size_t)pix * FF;
      for (int cc = 0; cc < FF; cc += 8) {
        v8h a1 = *(const v8h*)(p1 + cc);
        v8h a2 = *(const v8h*)(p2 + cc);
#pragma unroll
        for (int j = 0; j < 8; ++j) {
          acc += w[1 + cc + j] * (float)a1[j];
          acc += w[65 + cc + j] * (float)a2[j];
        }
      }
    }
  }
  out[i] = 1.f / (1.f + expf(-acc));
}

// ---------------- host launcher ----------------

extern "C" void kernel_launch(void* const* d_in, const int* in_sizes, int n_in,
                              void* d_out, int out_size, void* d_ws,
                              size_t ws_size, hipStream_t stream) {
  const float* img = (const float*)d_in[0];
  const float* wx1 = (const float*)d_in[1];
  const float* wh1 = (const float*)d_in[2];
  const float* b1 = (const float*)d_in[3];
  const float* g1 = (const float*)d_in[4];
  const float* be1 = (const float*)d_in[5];
  const float* m1 = (const float*)d_in[6];
  const float* v1 = (const float*)d_in[7];
  const float* wx2 = (const float*)d_in[8];
  const float* wh2 = (const float*)d_in[9];
  const float* b2 = (const float*)d_in[10];
  const float* g2 = (const float*)d_in[11];
  const float* be2 = (const float*)d_in[12];
  const float* m2 = (const float*)d_in[13];
  const float* v2 = (const float*)d_in[14];
  const float* w3 = (const float*)d_in[15];
  const float* b3 = (const float*)d_in[16];

  const size_t NPIX = (size_t)BATCH * HH * WW;          // 65536
  const size_t NSTATE = NPIX * FF;                      // 4194304
  const size_t NSEQ = (size_t)BATCH * TT * HH * WW;     // 655360

  float* out = (float*)d_out;
  float* c0 = out + NSEQ;
  float* c1 = c0 + NSTATE;
  float* h0 = c1 + NSTATE;
  float* h1 = h0 + NSTATE;

  char* wp = (char*)d_ws;
  auto alloc = [&](size_t bytes) -> char* {
    char* rv = wp;
    wp += (bytes + 255) & ~(size_t)255;
    return rv;
  };
  _Float16* wtx1 = (_Float16*)alloc(256 * 32 * 2);
  _Float16* wth1 = (_Float16*)alloc((size_t)25 * 256 * 64 * 2);
  _Float16* wtx2 = (_Float16*)alloc((size_t)9 * 256 * 64 * 2);
  _Float16* wth2 = (_Float16*)alloc((size_t)9 * 256 * 64 * 2);
  _Float16* h1A = (_Float16*)alloc(NSTATE * 2);
  _Float16* h1B = (_Float16*)alloc(NSTATE * 2);
  _Float16* h2A = (_Float16*)alloc(NSTATE * 2);
  _Float16* h2B = (_Float16*)alloc(NSTATE * 2);
  _Float16* x1s = (_Float16*)alloc(NSEQ * FF * 2);
  _Float16* x2s = (_Float16*)alloc(NSEQ * FF * 2);

  // weight repack (f32 HWIO -> f16 [tap][cout][cin] B-fragment layout)
  repack_oi_kernel<<<(25 * 64 * 256 + 255) / 256, 256, 0, stream>>>(
      wh1, wth1, 25, 64, 256);
  repack_oi_kernel<<<(9 * 64 * 256 + 255) / 256, 256, 0, stream>>>(
      wx2, wtx2, 9, 64, 256);
  repack_oi_kernel<<<(9 * 64 * 256 + 255) / 256, 256, 0, stream>>>(
      wh2, wth2, 9, 64, 256);
  repack_x1_kernel<<<32, 256, 0, stream>>>(wx1, wtx1);

  // init recurrent state
  int zb = (int)((NSTATE + 255) / 256);
  zero_f16_kernel<<<zb, 256, 0, stream>>>(h1A, (int)NSTATE);
  zero_f16_kernel<<<zb, 256, 0, stream>>>(h2A, (int)NSTATE);
  zero_f32_kernel<<<zb, 256, 0, stream>>>(c0, (int)NSTATE);
  zero_f32_kernel<<<zb, 256, 0, stream>>>(c1, (int)NSTATE);

  const int tiles = BATCH * HH * 2;  // 2048 workgroups per step

  for (int t = 0; t < TT; ++t) {
    const _Float16* hin = (t & 1) ? h1B : h1A;
    _Float16* hout = (t & 1) ? h1A : h1B;
    lstm1_step_kernel<<<tiles, 256, 0, stream>>>(
        img, t, wtx1, wth1, b1, g1, be1, m1, v1, hin, hout, c0, h0, x1s,
        t == TT - 1);
  }
  for (int t = 0; t < TT; ++t) {
    const _Float16* hin = (t & 1) ? h2B : h2A;
    _Float16* hout = (t & 1) ? h2A : h2B;
    lstm2_step_kernel<<<tiles, 256, 0, stream>>>(
        x1s, t, wtx2, wth2, b2, g2, be2, m2, v2, hin, hout, c1, h1, x2s,
        t == TT - 1);
  }
  head_kernel<<<(int)((NSEQ + 255) / 256), 256, 0, stream>>>(img, x1s, x2s, w3,
                                                             b3, out);
}
